// SaTformer_85693187490060
// MI455X (gfx1250) — matrix operand order repaired and checked by reading, not compiled, not run on hardware
//
#include <hip/hip_runtime.h>
#include <hip/hip_bf16.h>

typedef __bf16 bf16_t;
typedef __attribute__((ext_vector_type(16))) __bf16 v16bf;
typedef __attribute__((ext_vector_type(8)))  __bf16 v8bf;
typedef __attribute__((ext_vector_type(8)))  float  v8f;
typedef unsigned int v4u __attribute__((ext_vector_type(4)));
typedef int          v8i __attribute__((ext_vector_type(8)));
typedef int          v4i __attribute__((ext_vector_type(4)));

#define B_     2
#define F_     16
#define C_     11
#define H_     32
#define W_     32
#define SEQ_   1025
#define SEQP_  1056      // 33*32: pad so K/V chunks and GEMM tiles are exact
#define DIM_   512
#define HEADS_ 8
#define DH_    64
#define QKVLD_ 1536      // 3*DIM
#define NCLS_  1000
#define DEPTH_ 8

#if defined(__AMDGCN__) && __has_builtin(__builtin_amdgcn_tensor_load_to_lds) && \
    __has_builtin(__builtin_amdgcn_s_wait_tensorcnt)
#define HAVE_TDM 1
typedef __attribute__((address_space(3))) void lds_void_t;
#endif

// Combine the two contiguous 8-element runs into a v16bf A/B fragment.
// 16-bit A-matrix 16x32 layout (ISA 7.12.2): fragment element e, lane-group g:
// e<8 -> K = e + 8g, e>=8 -> K = 16 + (e-8) + 8g  (two contiguous 16B runs).
__device__ __forceinline__ v16bf frag_cat(v8bf lo, v8bf hi) {
    return __builtin_shufflevector(lo, hi, 0, 1, 2, 3, 4, 5, 6, 7,
                                           8, 9, 10, 11, 12, 13, 14, 15);
}

// ---------------------------------------------------------------------------
// Patch embedding + cls token + pos emb; zero-fills pad rows [SEQ, SEQP).
// Division-free gather loops.
// ---------------------------------------------------------------------------
__global__ __launch_bounds__(256) void patch_embed(
    const float* __restrict__ video, const float* __restrict__ pW,
    const float* __restrict__ pb, const float* __restrict__ cls,
    const float* __restrict__ pos, float* __restrict__ x)
{
    int idx = blockIdx.x * 256 + threadIdx.x;
    const int total = B_ * SEQP_ * DIM_;
    if (idx >= total) return;
    int d = idx & (DIM_ - 1);
    int t = (idx / DIM_) % SEQP_;
    int b = idx / (DIM_ * SEQP_);
    float out;
    if (t >= SEQ_) {
        out = 0.f;
    } else if (t == 0) {
        out = cls[d] + pos[d];
    } else {
        int tp = t - 1;
        int f  = tp >> 6;
        int hp = (tp >> 3) & 7;
        int wp = tp & 7;
        const float* vbase = video + ((size_t)b * F_ + f) * (C_ * H_ * W_)
                             + (hp * 4) * W_ + wp * 4;
        const float* wcol = pW + d;
        float acc = pb[d];
        for (int p1 = 0; p1 < 4; p1++) {
            for (int p2 = 0; p2 < 4; p2++) {
                const float* vp = vbase + p1 * W_ + p2;
                #pragma unroll
                for (int c = 0; c < C_; c++) {
                    acc += vp[(size_t)c * (H_ * W_)] * wcol[0];
                    wcol += DIM_;
                }
            }
        }
        out = acc + pos[(size_t)t * DIM_ + d];
    }
    x[idx] = out;
}

// ---------------------------------------------------------------------------
// LayerNorm (one wave per 512-wide row) -> bf16; zeroes pad rows.
// ---------------------------------------------------------------------------
__global__ __launch_bounds__(256) void ln_bf16(
    const float* __restrict__ x, const float* __restrict__ g,
    const float* __restrict__ bt, bf16_t* __restrict__ xn)
{
    const int tid = threadIdx.x;
    const int w = tid >> 5, lane = tid & 31;
    int row = blockIdx.x * 8 + w;
    if (row >= B_ * SEQP_) return;
    int t = row % SEQP_;
    bf16_t* dst = xn + (size_t)row * DIM_ + lane * 16;
    if (t >= SEQ_) {
        v8bf z;
        #pragma unroll
        for (int i = 0; i < 8; i++) z[i] = (bf16_t)0.f;
        *(v8bf*)dst = z;
        *(v8bf*)(dst + 8) = z;
        return;
    }
    const float* src = x + (size_t)row * DIM_ + lane * 16;
    float4 va = ((const float4*)src)[0];
    float4 vb = ((const float4*)src)[1];
    float4 vc = ((const float4*)src)[2];
    float4 vd = ((const float4*)src)[3];
    float v[16] = {va.x, va.y, va.z, va.w, vb.x, vb.y, vb.z, vb.w,
                   vc.x, vc.y, vc.z, vc.w, vd.x, vd.y, vd.z, vd.w};
    float s = 0.f, ss = 0.f;
    #pragma unroll
    for (int i = 0; i < 16; i++) { s += v[i]; ss += v[i] * v[i]; }
    #pragma unroll
    for (int off = 1; off < 32; off <<= 1) {
        s  += __shfl_xor(s,  off, 32);
        ss += __shfl_xor(ss, off, 32);
    }
    float mean = s * (1.f / DIM_);
    float var  = ss * (1.f / DIM_) - mean * mean;
    float rstd = rsqrtf(var + 1e-5f);
    const float* gp = g + lane * 16;
    const float* bp = bt + lane * 16;
    v8bf o0, o1;
    #pragma unroll
    for (int i = 0; i < 8; i++)
        o0[i] = (bf16_t)((v[i] - mean) * rstd * gp[i] + bp[i]);
    #pragma unroll
    for (int i = 0; i < 8; i++)
        o1[i] = (bf16_t)((v[8 + i] - mean) * rstd * gp[8 + i] + bp[8 + i]);
    *(v8bf*)dst = o0;
    *(v8bf*)(dst + 8) = o1;
}

// ---------------------------------------------------------------------------
// WMMA GEMM: C[M,N](f32) = A_bf16[M,K] @ W_f32->bf16[K,N] (+bias+resid)
// Block tile 64x64, 8 waves as 4(M) x 2(N); each wave: one A-frag row,
// two 16x16 accumulators.  A tile moved global->LDS by the Tensor Data
// Mover when available; W tile converted f32->bf16 into a transposed LDS
// tile so B fragments are contiguous 16B LDS loads.  Exact tiling
// (M % 64 == 0, N % 64 == 0, K % 32 == 0 for all uses here).
// ---------------------------------------------------------------------------
template <bool EPILOGUE>
__global__ __launch_bounds__(256) void gemm_bf16_wmma(
    const bf16_t* __restrict__ A, const float* __restrict__ Wt,
    float* __restrict__ Cout, const float* __restrict__ bias,
    const float* __restrict__ resid, int M, int N, int K)
{
    __shared__ bf16_t As[64][32];        // row-major A tile (64 rows x 32 k)
    __shared__ bf16_t BsT[64][40];       // transposed W tile: [n][k], padded
    const int tid  = threadIdx.x;
    const int lane = tid & 31, wave = tid >> 5;
    const int lg = lane >> 4, lnn = lane & 15;
    const int wm = wave >> 1;            // 0..3 : 16-row M subtile
    const int wn = wave & 1;             // 0..1 : 32-col N subtile
    const int bm = blockIdx.y * 64;
    const int bn = blockIdx.x * 64;

    v8f acc[2];
    #pragma unroll
    for (int nt = 0; nt < 2; nt++)
        #pragma unroll
        for (int s = 0; s < 8; s++) acc[nt][s] = 0.f;

    // W-tile fill mapping: column strip per thread (coalesced across lanes)
    const int fn  = tid & 63;            // n within tile
    const int fk0 = (tid >> 6) * 8;      // 8 consecutive k per thread

#ifdef HAVE_TDM
    const unsigned ldsAs = (unsigned)(unsigned long long)(lds_void_t*)(&As[0][0]);
#endif

    for (int k0 = 0; k0 < K; k0 += 32) {
        // ---- A tile: global (bf16) -> LDS ----
#ifdef HAVE_TDM
        if (tid < 32) {                  // wave 0 only (wave-uniform branch)
            unsigned long long ga =
                (unsigned long long)(const void*)(A + (size_t)bm * K + k0);
            v4u g0; v8i g1; v4i gz4;
            gz4[0] = 0; gz4[1] = 0; gz4[2] = 0; gz4[3] = 0;
            g0[0] = 1u;                                   // count=1, user D#
            g0[1] = ldsAs;                                // lds_addr
            g0[2] = (unsigned)(ga & 0xffffffffu);         // global_addr lo
            g0[3] = (unsigned)((ga >> 32) & 0x1ffffffu)   // global_addr hi
                    | (2u << 30);                         // type=2 (image)
            g1[0] = (int)(1u << 16);                      // data_size = 2B
            g1[1] = (int)(32u << 16);                     // tensor_dim0 = 32
            g1[2] = (int)(64u << 16);                     // tensor_dim1 = 64
            g1[3] = (int)(32u << 16);                     // tile_dim0 = 32
            g1[4] = (int)64u;                             // tile_dim1 = 64
            g1[5] = (int)(unsigned)K;                     // dim0 stride = K
            g1[6] = 0; g1[7] = 0;
#if __clang_major__ >= 23
            v8i gz8;
            #pragma unroll
            for (int i = 0; i < 8; i++) gz8[i] = 0;
            __builtin_amdgcn_tensor_load_to_lds(g0, g1, gz4, gz4, gz8, 0);
#else
            __builtin_amdgcn_tensor_load_to_lds(g0, g1, gz4, gz4, 0);
#endif
            __builtin_amdgcn_s_wait_tensorcnt(0);
        }
#else
        {
            int r  = tid >> 2;
            int c0 = (tid & 3) * 8;
            *(v8bf*)&As[r][c0] = *(const v8bf*)(A + (size_t)(bm + r) * K + k0 + c0);
        }
#endif
        // ---- W tile: f32 -> bf16, transposed into LDS ----
        {
            v8bf pk;
            #pragma unroll
            for (int i = 0; i < 8; i++) {
                const float* src = Wt + (size_t)(k0 + fk0 + i) * N + bn + fn;
                pk[i] = (bf16_t)src[0];
                __builtin_prefetch(src + 32 * (size_t)N, 0, 1);  // next k-step
            }
            *(v8bf*)&BsT[fn][fk0] = pk;
        }
        __syncthreads();

        // ---- fragments (two contiguous 16B LDS loads each) + 2 WMMAs ----
        const v8bf* arow = (const v8bf*)&As[wm * 16 + lnn][0];
        v16bf af = frag_cat(arow[lg], arow[2 + lg]);
        #pragma unroll
        for (int nt = 0; nt < 2; nt++) {
            const v8bf* brow = (const v8bf*)&BsT[wn * 32 + nt * 16 + lnn][0];
            v16bf bf = frag_cat(brow[lg], brow[2 + lg]);
            acc[nt] = __builtin_amdgcn_wmma_f32_16x16x32_bf16(
                          false, af, false, bf, (short)0, acc[nt], false, false);
        }
        __syncthreads();
    }

    // ---- epilogue: exact tiling, compile-time bias/resid specialization ----
    #pragma unroll
    for (int nt = 0; nt < 2; nt++) {
        const int col = bn + wn * 32 + nt * 16 + lnn;
        const float bcol = EPILOGUE ? bias[col] : 0.f;
        #pragma unroll
        for (int s = 0; s < 8; s++) {
            int row = bm + wm * 16 + s + 8 * lg;   // C layout: M = s + 8*(lane>>4)
            float v = acc[nt][s] + bcol;
            if (EPILOGUE) v += resid[(size_t)row * N + col];
            Cout[(size_t)row * N + col] = v;
        }
    }
}

// ---------------------------------------------------------------------------
// Flash attention: one wave per (b, head, 16-row q block), j chunks of 32.
// QK^T and P@V via v_wmma_f32_16x16x32_bf16; online softmax in C layout.
// ---------------------------------------------------------------------------
__global__ __launch_bounds__(256) void attn_wmma(
    const float* __restrict__ qkv, bf16_t* __restrict__ obf)
{
    __shared__ bf16_t Plds[8][16][32];      // per-wave P transpose staging
    const int tid = threadIdx.x;
    const int w = tid >> 5, lane = tid & 31;
    const int lg = lane >> 4, lnn = lane & 15;
    const int QB = (SEQ_ + 15) / 16;        // 65
    int wid = blockIdx.x * 8 + w;
    if (wid >= B_ * HEADS_ * QB) return;
    const int qb = wid % QB;
    const int h  = (wid / QB) % HEADS_;
    const int b  = wid / (QB * HEADS_);

    const float scale = 0.125f;             // DH^-0.5
    const size_t baseRow = (size_t)b * SEQP_;

    // Q fragments (scaled at load); two contiguous float4-pair runs per frag
    v16bf aQ[2];
    {
        const float* qrow = qkv + (baseRow + qb * 16 + lnn) * QKVLD_ + h * DH_;
        #pragma unroll
        for (int kc = 0; kc < 2; kc++) {
            const float4* lo = (const float4*)(qrow + kc * 32 + 16 * lg);
            const float4* hi = (const float4*)(qrow + kc * 32 + 16 * lg + 8);
            float4 l0 = lo[0], l1 = lo[1], h0 = hi[0], h1 = hi[1];
            float lv[8] = {l0.x, l0.y, l0.z, l0.w, l1.x, l1.y, l1.z, l1.w};
            float hv[8] = {h0.x, h0.y, h0.z, h0.w, h1.x, h1.y, h1.z, h1.w};
            #pragma unroll
            for (int e = 0; e < 8; e++) {
                aQ[kc][e]     = (bf16_t)(lv[e] * scale);
                aQ[kc][8 + e] = (bf16_t)(hv[e] * scale);
            }
        }
    }

    float mrun[8], lrun[8];
    v8f accO[4];
    #pragma unroll
    for (int s = 0; s < 8; s++) { mrun[s] = -1e30f; lrun[s] = 0.f; }
    #pragma unroll
    for (int t = 0; t < 4; t++)
        #pragma unroll
        for (int s = 0; s < 8; s++) accO[t][s] = 0.f;

    for (int j0 = 0; j0 < SEQ_; j0 += 32) {
        // ---- sim = Q @ K^T for 32 columns (two 16x16 tiles) ----
        v8f sim[2];
        #pragma unroll
        for (int nt = 0; nt < 2; nt++) {
            #pragma unroll
            for (int s = 0; s < 8; s++) sim[nt][s] = 0.f;
            const float* krow = qkv + (baseRow + j0 + nt * 16 + lnn) * QKVLD_
                                + DIM_ + h * DH_;
            #pragma unroll
            for (int kc = 0; kc < 2; kc++) {
                const float4* kp = (const float4*)(krow + kc * 32 + 16 * lg);
                float4 k0v = kp[0], k1v = kp[1], k2v = kp[2], k3v = kp[3];
                float kv[16] = {k0v.x, k0v.y, k0v.z, k0v.w, k1v.x, k1v.y, k1v.z, k1v.w,
                                k2v.x, k2v.y, k2v.z, k2v.w, k3v.x, k3v.y, k3v.z, k3v.w};
                v16bf bK;               // B layout: lane = column, K = 16*lg + e
                #pragma unroll
                for (int e = 0; e < 16; e++) bK[e] = (bf16_t)kv[e];
                sim[nt] = __builtin_amdgcn_wmma_f32_16x16x32_bf16(
                              false, aQ[kc], false, bK, (short)0, sim[nt], false, false);
            }
            if (j0 + nt * 16 + lnn >= SEQ_) {   // mask out-of-range columns
                #pragma unroll
                for (int s = 0; s < 8; s++) sim[nt][s] = -1e30f;
            }
        }
        // ---- online softmax (8 rows in slots; reduce across 16 lanes) ----
        float fac[8];
        #pragma unroll
        for (int s = 0; s < 8; s++) {
            float mx = fmaxf(sim[0][s], sim[1][s]);
            #pragma unroll
            for (int off = 1; off < 16; off <<= 1)
                mx = fmaxf(mx, __shfl_xor(mx, off, 32));
            float mnew = fmaxf(mrun[s], mx);
            float f  = __expf(mrun[s] - mnew);
            float p0 = __expf(sim[0][s] - mnew);
            float p1 = __expf(sim[1][s] - mnew);
            sim[0][s] = p0; sim[1][s] = p1;
            float sum = p0 + p1;
            #pragma unroll
            for (int off = 1; off < 16; off <<= 1)
                sum += __shfl_xor(sum, off, 32);
            lrun[s] = lrun[s] * f + sum;
            mrun[s] = mnew;
            fac[s]  = f;
        }
        #pragma unroll
        for (int t = 0; t < 4; t++)
            #pragma unroll
            for (int s = 0; s < 8; s++) accO[t][s] *= fac[s];

        // ---- transpose P (C layout -> A layout) through per-wave LDS ----
        #pragma unroll
        for (int s = 0; s < 8; s++) {
            Plds[w][s + 8 * lg][lnn]      = (bf16_t)sim[0][s];
            Plds[w][s + 8 * lg][16 + lnn] = (bf16_t)sim[1][s];
        }
        asm volatile("s_wait_dscnt 0x0" ::: "memory");  // LDS in-order per wave
        const v8bf* pr = (const v8bf*)&Plds[w][lnn][0];
        v16bf aP = frag_cat(pr[2 * lg], pr[2 * lg + 1]);

        // ---- accO += P @ V (four 16-wide DH tiles) ----
        #pragma unroll
        for (int t = 0; t < 4; t++) {
            v16bf bV;
            const float* vcol = qkv + (baseRow + j0 + 16 * lg) * QKVLD_
                                + 2 * DIM_ + h * DH_ + t * 16 + lnn;
            #pragma unroll
            for (int e = 0; e < 16; e++)
                bV[e] = (bf16_t)vcol[(size_t)e * QKVLD_];
            accO[t] = __builtin_amdgcn_wmma_f32_16x16x32_bf16(
                          false, aP, false, bV, (short)0, accO[t], false, false);
        }
    }

    // ---- write O / l as bf16 for the out-projection GEMM ----
    #pragma unroll
    for (int s = 0; s < 8; s++) {
        int trow = qb * 16 + s + 8 * lg;
        if (trow < SEQ_) {
            float inv = 1.f / lrun[s];
            #pragma unroll
            for (int t = 0; t < 4; t++)
                obf[(baseRow + trow) * (size_t)DIM_ + h * DH_ + t * 16 + lnn]
                    = (bf16_t)(accO[t][s] * inv);
        }
    }
}

// ---------------------------------------------------------------------------
// Final: LN(cls token) @ cls_W + cls_b   (2 x 1000, trivial)
// ---------------------------------------------------------------------------
__global__ __launch_bounds__(256) void cls_head(
    const float* __restrict__ x, const float* __restrict__ g,
    const float* __restrict__ bt, const float* __restrict__ Wc,
    const float* __restrict__ bc, float* __restrict__ out)
{
    __shared__ float xs[DIM_];
    __shared__ float red[256];
    const int tid = threadIdx.x;
    const int b = blockIdx.x;
    const float* row = x + (size_t)b * SEQP_ * DIM_;   // token 0 of batch b
    float v0 = row[tid], v1 = row[tid + 256];
    red[tid] = v0 + v1;
    __syncthreads();
    for (int off = 128; off > 0; off >>= 1) {
        if (tid < off) red[tid] += red[tid + off];
        __syncthreads();
    }
    float mean = red[0] * (1.f / DIM_);
    __syncthreads();
    red[tid] = (v0 - mean) * (v0 - mean) + (v1 - mean) * (v1 - mean);
    __syncthreads();
    for (int off = 128; off > 0; off >>= 1) {
        if (tid < off) red[tid] += red[tid + off];
        __syncthreads();
    }
    float rstd = rsqrtf(red[0] * (1.f / DIM_) + 1e-5f);
    xs[tid]       = (v0 - mean) * rstd * g[tid]       + bt[tid];
    xs[tid + 256] = (v1 - mean) * rstd * g[tid + 256] + bt[tid + 256];
    __syncthreads();
    for (int j = tid; j < NCLS_; j += 256) {
        float acc = bc[j];
        for (int k = 0; k < DIM_; k++) acc += xs[k] * Wc[(size_t)k * NCLS_ + j];
        out[(size_t)b * NCLS_ + j] = acc;
    }
}

// ---------------------------------------------------------------------------
extern "C" void kernel_launch(void* const* d_in, const int* in_sizes, int n_in,
                              void* d_out, int out_size, void* d_ws, size_t ws_size,
                              hipStream_t stream)
{
    (void)in_sizes; (void)n_in; (void)out_size; (void)ws_size;
    const float* video  = (const float*)d_in[0];
    const float* patchW = (const float*)d_in[1];
    const float* patchb = (const float*)d_in[2];
    const float* clstok = (const float*)d_in[3];
    const float* posemb = (const float*)d_in[4];
    const float* ln_g   = (const float*)d_in[5];
    const float* ln_b   = (const float*)d_in[6];
    const float* Wqkv   = (const float*)d_in[7];
    const float* Wout   = (const float*)d_in[8];
    const float* bout   = (const float*)d_in[9];
    const float* oln_g  = (const float*)d_in[10];
    const float* oln_b  = (const float*)d_in[11];
    const float* clsW   = (const float*)d_in[12];
    const float* clsb   = (const float*)d_in[13];
    float* outp = (float*)d_out;

    char* ws = (char*)d_ws;
    auto alignup = [](size_t v) { return (v + 255) & ~(size_t)255; };
    const size_t ROWS = (size_t)B_ * SEQP_;          // 2112 = 33 * 64
    float*  x   = (float*)ws;   ws += alignup(ROWS * DIM_   * sizeof(float));
    float*  qkv = (float*)ws;   ws += alignup(ROWS * QKVLD_ * sizeof(float));
    bf16_t* xn  = (bf16_t*)ws;  ws += alignup(ROWS * DIM_   * sizeof(bf16_t));
    bf16_t* ob  = (bf16_t*)ws;  ws += alignup(ROWS * DIM_   * sizeof(bf16_t));

    {
        int total = B_ * SEQP_ * DIM_;
        patch_embed<<<(total + 255) / 256, 256, 0, stream>>>(
            video, patchW, patchb, clstok, posemb, x);
    }

    const int M = (int)ROWS;                         // 2112
    for (int l = 0; l < DEPTH_; l++) {
        ln_bf16<<<(M + 7) / 8, 256, 0, stream>>>(
            x, ln_g + l * DIM_, ln_b + l * DIM_, xn);

        dim3 gq(3 * DIM_ / 64, M / 64);              // 24 x 33
        gemm_bf16_wmma<false><<<gq, 256, 0, stream>>>(
            xn, Wqkv + (size_t)l * DIM_ * 3 * DIM_, qkv,
            nullptr, nullptr, M, 3 * DIM_, DIM_);

        int waves = B_ * HEADS_ * ((SEQ_ + 15) / 16);   // 1040
        attn_wmma<<<(waves + 7) / 8, 256, 0, stream>>>(qkv, ob);

        dim3 go(DIM_ / 64, M / 64);                  // 8 x 33
        gemm_bf16_wmma<true><<<go, 256, 0, stream>>>(
            ob, Wout + (size_t)l * DIM_ * DIM_, x,
            bout + l * DIM_, x, M, DIM_, DIM_);
    }

    cls_head<<<B_, 256, 0, stream>>>(x, oln_g, oln_b, clsW, clsb, outp);
}
